// TKConv2dR_75127567942280
// MI455X (gfx1250) — matrix operand order repaired and checked by reading, not compiled, not run on hardware
//
#include <hip/hip_runtime.h>
#include <hip/hip_bf16.h>

// Tucker-factorized conv2d for MI455X (gfx1250, wave32, WMMA bf16):
//   prepack: first/last/core -> bf16 (core relaid as [tap][p][q])
//   y1 = first (128x512) * x_b (512x784)          [per batch]
//   y2 = conv3x3(y1, core) as implicit GEMM        K = 128*9
//   out = last (512x128) * y2 + bias
// GEMMs: 128(M) x 64(N) block tiles, K-step 32, 8 waves/block,
// bf16 A/B fragments + f32 accumulation via v_wmma_f32_16x16x32_bf16.

typedef __attribute__((ext_vector_type(16))) __bf16 v16bf;
typedef __attribute__((ext_vector_type(8)))  __bf16 v8bf;
typedef __attribute__((ext_vector_type(8)))  float  v8f;

#define LSTR 40           // LDS row stride (bf16 elems): 80B rows, 16B-aligned
#define NPIX 784          // 28*28
#define NT   13           // ceil(784/64) N-tiles

struct BFrag { uint4 lo, hi; };

// A-matrix 16x32 bf16 fragment (ISA 7.12.2): lane m=L%16; khalf=(L>=16).
// elems 0..7  -> K = 8*khalf + 0..7 ; elems 8..15 -> K = 16 + 8*khalf + 0..7
__device__ __forceinline__ v16bf frag_a(const unsigned short* lds, int row, int khalf) {
  const unsigned short* p = lds + row * LSTR + khalf * 8;
  BFrag t;
  t.lo = *reinterpret_cast<const uint4*>(p);
  t.hi = *reinterpret_cast<const uint4*>(p + 16);
  return __builtin_bit_cast(v16bf, t);
}

// B-matrix 32x16 bf16 fragment: lane n=L%16 holds K = 8*khalf + 0..7, 16 + 8*khalf + 0..7
// (LDS holds B transposed: Bt[n][k], so the 16 K values are contiguous)
__device__ __forceinline__ v16bf frag_b(const unsigned short* lds, int row, int khalf) {
  const unsigned short* p = lds + row * LSTR + khalf * 8;
  BFrag t;
  t.lo = *reinterpret_cast<const uint4*>(p);
  t.hi = *reinterpret_cast<const uint4*>(p + 16);
  return __builtin_bit_cast(v16bf, t);
}

__device__ __forceinline__ v8f vzero8() {
  v8f z;
#pragma unroll
  for (int e = 0; e < 8; ++e) z[e] = 0.0f;
  return z;
}

__device__ __forceinline__ unsigned short bfbits(__bf16 h) {
  return __builtin_bit_cast(unsigned short, h);
}

// Copy 16 contiguous bf16 (32B) from global to LDS as 2 x b128
__device__ __forceinline__ void copy16(unsigned short* dst, const unsigned short* src) {
  *reinterpret_cast<uint4*>(dst)     = *reinterpret_cast<const uint4*>(src);
  *reinterpret_cast<uint4*>(dst + 8) = *reinterpret_cast<const uint4*>(src + 8);
}

// Preload A + all 4 B fragments, then issue the 4 WMMAs (lets the compiler
// batch the 10 ds_load_b128 and relax s_wait_dscnt instead of 0-waits per WMMA)
__device__ __forceinline__ void mma_step(const unsigned short* As, const unsigned short* Bs,
                                         int wave, int l16, int khalf, v8f acc[4]) {
  v16bf a  = frag_a(As, wave * 16 + l16, khalf);
  v16bf b0 = frag_b(Bs,  0 + l16, khalf);
  v16bf b1 = frag_b(Bs, 16 + l16, khalf);
  v16bf b2 = frag_b(Bs, 32 + l16, khalf);
  v16bf b3 = frag_b(Bs, 48 + l16, khalf);
  acc[0] = __builtin_amdgcn_wmma_f32_16x16x32_bf16(false, a, false, b0, (short)0, acc[0], false, false);
  acc[1] = __builtin_amdgcn_wmma_f32_16x16x32_bf16(false, a, false, b1, (short)0, acc[1], false, false);
  acc[2] = __builtin_amdgcn_wmma_f32_16x16x32_bf16(false, a, false, b2, (short)0, acc[2], false, false);
  acc[3] = __builtin_amdgcn_wmma_f32_16x16x32_bf16(false, a, false, b3, (short)0, acc[3], false, false);
}

// ---------------- Prepack: f32 weights -> bf16 (core -> [tap][p][q]) ----------------
// firstb: 128*512, lastb: 512*128, coreb: 9*128*128 (tap-major)
__global__ __launch_bounds__(256) void tk_prepack(
    const float* __restrict__ first, const float* __restrict__ last,
    const float* __restrict__ core,
    unsigned short* __restrict__ firstb, unsigned short* __restrict__ lastb,
    unsigned short* __restrict__ coreb)
{
  const int e0 = (blockIdx.x * 256 + threadIdx.x) * 8;
  v8f f;
  unsigned short* d;
  if (e0 < 65536) {                       // first: 128*512
    const float* s = first + e0;
#pragma unroll
    for (int j = 0; j < 8; ++j) f[j] = s[j];
    d = firstb + e0;
  } else if (e0 < 131072) {               // last: 512*128
    const float* s = last + (e0 - 65536);
#pragma unroll
    for (int j = 0; j < 8; ++j) f[j] = s[j];
    d = lastb + (e0 - 65536);
  } else {                                // core -> [tap][p][q]
    const int ep  = e0 - 131072;          // output idx: tap*16384 + p*128 + q
    const int tap = ep >> 14;
    const int r   = ep & 16383;
    const int p   = r >> 7, q = r & 127;
    const float* s = core + ((size_t)p * 128 + q) * 9 + tap;  // stride 9 along q
#pragma unroll
    for (int j = 0; j < 8; ++j) f[j] = s[j * 9];
    d = coreb + ep;
  }
  v8bf h = __builtin_convertvector(f, v8bf);
#pragma unroll
  for (int j = 0; j < 8; ++j) d[j] = bfbits(h[j]);
}

// ---------------- Stage 1: y1[b,q,pix] = sum_i first[q,i] * x[b,i,pix] ----------------
__global__ __launch_bounds__(256) void tk_stage1(
    const float* __restrict__ x, const unsigned short* __restrict__ firstb,
    unsigned short* __restrict__ y1)
{
  __shared__ __attribute__((aligned(16))) unsigned short As[128 * LSTR];
  __shared__ __attribute__((aligned(16))) unsigned short Bs[64 * LSTR];

  const int b   = blockIdx.y;
  const int n0  = blockIdx.x * 64;
  const int tid = threadIdx.x;
  const int lane = tid & 31, wave = tid >> 5;
  const int khalf = lane >> 4, l16 = lane & 15;

  v8f acc[4];
#pragma unroll
  for (int i = 0; i < 4; ++i) acc[i] = vzero8();

  const float* xb = x + (size_t)b * (512 * NPIX);
  const int kB = tid >> 3, ncB = (tid & 7) * 8;
  // clamp per-element pixel index in-bounds; mask out-of-range with 0
  int  npix[8];
  bool nok[8];
#pragma unroll
  for (int j = 0; j < 8; ++j) {
    int n = n0 + ncB + j;
    nok[j]  = n < NPIX;
    npix[j] = nok[j] ? n : (NPIX - 1);
  }

  for (int k0 = 0; k0 < 512; k0 += 32) {
    __syncthreads();
    { // A tile: firstb[0..127][k0..k0+31] (bf16, contiguous)
      int row = tid >> 1, kc = (tid & 1) * 16;
      copy16(As + row * LSTR + kc, firstb + row * 512 + k0 + kc);
    }
    { // B tile (transposed): Bt[n][k] = bf16(x[b][k0+k][n])
      const float* s = xb + (size_t)(k0 + kB) * NPIX;
      v8f f;
#pragma unroll
      for (int j = 0; j < 8; ++j) {
        float v = s[npix[j]];
        f[j] = nok[j] ? v : 0.0f;
      }
      v8bf h = __builtin_convertvector(f, v8bf);   // v_cvt_pk_bf16_f32
#pragma unroll
      for (int j = 0; j < 8; ++j) Bs[(ncB + j) * LSTR + kB] = bfbits(h[j]);
    }
    __syncthreads();
    mma_step(As, Bs, wave, l16, khalf, acc);
  }

  unsigned short* y1b = y1 + (size_t)b * (128 * NPIX);
#pragma unroll
  for (int ns = 0; ns < 4; ++ns) {
    int n = n0 + ns * 16 + l16;
    if (n < NPIX) {
#pragma unroll
      for (int r = 0; r < 8; ++r) {
        int q = wave * 16 + khalf * 8 + r;
        y1b[q * NPIX + n] = bfbits((__bf16)acc[ns][r]);
      }
    }
  }
}

// ---------------- Stage 2: y2 = conv3x3(y1, core), implicit GEMM (K = 9*128) ----------------
__global__ __launch_bounds__(256) void tk_stage2(
    const unsigned short* __restrict__ y1, const unsigned short* __restrict__ coreb,
    unsigned short* __restrict__ y2)
{
  __shared__ __attribute__((aligned(16))) unsigned short As[128 * LSTR];
  __shared__ __attribute__((aligned(16))) unsigned short Bs[64 * LSTR];

  const int b   = blockIdx.y;
  const int n0  = blockIdx.x * 64;
  const int tid = threadIdx.x;
  const int lane = tid & 31, wave = tid >> 5;
  const int khalf = lane >> 4, l16 = lane & 15;

  v8f acc[4];
#pragma unroll
  for (int i = 0; i < 4; ++i) acc[i] = vzero8();

  const unsigned short* y1b = y1 + (size_t)b * (128 * NPIX);

  // Hoist per-thread pixel coordinates for the B-tile gather
  const int kB = tid >> 3, ncB = (tid & 7) * 8;
  int ph[8], pw[8];
  bool pv[8];
#pragma unroll
  for (int j = 0; j < 8; ++j) {
    int pix = n0 + ncB + j;
    pv[j] = pix < NPIX;
    ph[j] = pix / 28;
    pw[j] = pix % 28;
  }

  for (int tap = 0; tap < 9; ++tap) {
    const int kh = tap / 3 - 1, kw = tap % 3 - 1;
    for (int qc = 0; qc < 128; qc += 32) {
      __syncthreads();
      { // A tile: coreb[tap][0..127][qc..qc+31] (bf16, contiguous)
        int row = tid >> 1, c = (tid & 1) * 16;
        copy16(As + row * LSTR + c, coreb + tap * 16384 + row * 128 + qc + c);
      }
      { // B tile: shifted y1 with zero padding (addresses clamped in-bounds)
        int q = qc + kB;
#pragma unroll
        for (int j = 0; j < 8; ++j) {
          int h2 = ph[j] + kh, w2 = pw[j] + kw;
          bool ok = pv[j] && ((unsigned)h2 < 28u) && ((unsigned)w2 < 28u);
          int off = ok ? (h2 * 28 + w2) : 0;
          unsigned short v = y1b[q * NPIX + off];
          if (!ok) v = 0;
          Bs[(ncB + j) * LSTR + kB] = v;
        }
      }
      __syncthreads();
      mma_step(As, Bs, wave, l16, khalf, acc);
    }
  }

  unsigned short* y2b = y2 + (size_t)b * (128 * NPIX);
#pragma unroll
  for (int ns = 0; ns < 4; ++ns) {
    int n = n0 + ns * 16 + l16;
    if (n < NPIX) {
#pragma unroll
      for (int r = 0; r < 8; ++r) {
        int p = wave * 16 + khalf * 8 + r;
        y2b[p * NPIX + n] = bfbits((__bf16)acc[ns][r]);
      }
    }
  }
}

// ---------------- Stage 3: out[b,o,pix] = sum_p last[o,p] * y2[b,p,pix] + bias[o] ----------------
__global__ __launch_bounds__(256) void tk_stage3(
    const unsigned short* __restrict__ y2, const unsigned short* __restrict__ lastb,
    const float* __restrict__ bias, float* __restrict__ out)
{
  __shared__ __attribute__((aligned(16))) unsigned short As[128 * LSTR];
  __shared__ __attribute__((aligned(16))) unsigned short Bs[64 * LSTR];

  const int b   = blockIdx.y;
  const int n0  = blockIdx.x * 64;
  const int m0  = blockIdx.z * 128;
  const int tid = threadIdx.x;
  const int lane = tid & 31, wave = tid >> 5;
  const int khalf = lane >> 4, l16 = lane & 15;

  v8f acc[4];
#pragma unroll
  for (int i = 0; i < 4; ++i) acc[i] = vzero8();

  const unsigned short* y2b = y2 + (size_t)b * (128 * NPIX);
  const int kB = tid >> 3, ncB = (tid & 7) * 8;
  int  npix[8];
  bool nok[8];
#pragma unroll
  for (int j = 0; j < 8; ++j) {
    int n = n0 + ncB + j;
    nok[j]  = n < NPIX;
    npix[j] = nok[j] ? n : (NPIX - 1);
  }

  for (int k0 = 0; k0 < 128; k0 += 32) {
    __syncthreads();
    { // A tile: lastb[m0+row][k0..k0+31] (bf16, contiguous)
      int row = tid >> 1, c = (tid & 1) * 16;
      copy16(As + row * LSTR + c, lastb + (size_t)(m0 + row) * 128 + k0 + c);
    }
    { // B tile: Bt[n][k] = y2[b][k0+k][n] (already bf16)
      const unsigned short* s = y2b + (size_t)(k0 + kB) * NPIX;
#pragma unroll
      for (int j = 0; j < 8; ++j) {
        unsigned short v = s[npix[j]];
        if (!nok[j]) v = 0;
        Bs[(ncB + j) * LSTR + kB] = v;
      }
    }
    __syncthreads();
    mma_step(As, Bs, wave, l16, khalf, acc);
  }

  float bv[8];
#pragma unroll
  for (int r = 0; r < 8; ++r) bv[r] = bias[m0 + wave * 16 + khalf * 8 + r];

  float* outb = out + (size_t)b * (512 * NPIX);
#pragma unroll
  for (int ns = 0; ns < 4; ++ns) {
    int n = n0 + ns * 16 + l16;
    if (n < NPIX) {
#pragma unroll
      for (int r = 0; r < 8; ++r) {
        int m = m0 + wave * 16 + khalf * 8 + r;
        outb[(size_t)m * NPIX + n] = acc[ns][r] + bv[r];
      }
    }
  }
}

extern "C" void kernel_launch(void* const* d_in, const int* in_sizes, int n_in,
                              void* d_out, int out_size, void* d_ws, size_t ws_size,
                              hipStream_t stream) {
  (void)in_sizes; (void)n_in; (void)out_size; (void)ws_size;
  const float* x     = (const float*)d_in[0];   // (64,512,28,28)
  const float* first = (const float*)d_in[1];   // (128,512)
  const float* core  = (const float*)d_in[2];   // (128,128,3,3)
  const float* last  = (const float*)d_in[3];   // (512,128)
  const float* bias  = (const float*)d_in[4];   // (512,)
  float* out = (float*)d_out;                   // (64,512,28,28)

  // Workspace layout (all 16B aligned):
  unsigned short* y1     = (unsigned short*)d_ws;             // 64*128*784
  unsigned short* y2     = y1 + (size_t)64 * 128 * NPIX;      // 64*128*784
  unsigned short* firstb = y2 + (size_t)64 * 128 * NPIX;      // 128*512
  unsigned short* lastb  = firstb + 128 * 512;                // 512*128
  unsigned short* coreb  = lastb + 512 * 128;                 // 9*128*128

  dim3 blk(256);
  tk_prepack<<<dim3(136),      blk, 0, stream>>>(first, last, core, firstb, lastb, coreb);
  tk_stage1 <<<dim3(NT, 64),   blk, 0, stream>>>(x, firstb, y1);
  tk_stage2 <<<dim3(NT, 64),   blk, 0, stream>>>(y1, coreb, y2);
  tk_stage3 <<<dim3(NT, 64, 4), blk, 0, stream>>>(y2, lastb, bias, out);
}
